// SparseMoE_24515673326234
// MI455X (gfx1250) — matrix operand order repaired and checked by reading, not compiled, not run on hardware
//
#include <hip/hip_runtime.h>
#include <hip/hip_bf16.h>

// ---------------- problem constants (from reference setup) ----------------
#define N_TOK 4096      // B*S = 2*2048
#define D_DIM 1024
#define H_DIM 4096
#define E_NUM 8
#define TOPK  2
#define MTILE 64        // rows per FFN block
#define RMAX  8704      // N_TOK*TOPK + E_NUM*64 (segment padding to 64)

typedef __attribute__((ext_vector_type(16))) __bf16 v16bf;
typedef __attribute__((ext_vector_type(8)))  float  v8f;

union AF {
    v16bf v;
    uint4 u4[2];
};

__device__ __forceinline__ unsigned short f2bf(float f) {
    union { float f; unsigned u; } v;
    v.f = f;
    unsigned r = v.u + 0x7FFFu + ((v.u >> 16) & 1u);   // round-to-nearest-even
    return (unsigned short)(r >> 16);
}

// CDNA5 async copy: 16B global -> LDS, tracked by ASYNCcnt (same IOFFSET applied
// to both the LDS and global addresses per the ISA pseudocode).
#define ASYNC_CP16(laddr, gaddr, OFF)                                          \
    asm volatile("global_load_async_to_lds_b128 %0, %1, off offset:" #OFF      \
                 :: "v"(laddr), "v"(gaddr) : "memory")

__device__ __forceinline__ void async_stage_row128(unsigned laddr, unsigned long long gaddr) {
    // 128 bytes = 8 x b128 async transfers per thread
    ASYNC_CP16(laddr, gaddr, 0);
    ASYNC_CP16(laddr, gaddr, 16);
    ASYNC_CP16(laddr, gaddr, 32);
    ASYNC_CP16(laddr, gaddr, 48);
    ASYNC_CP16(laddr, gaddr, 64);
    ASYNC_CP16(laddr, gaddr, 80);
    ASYNC_CP16(laddr, gaddr, 96);
    ASYNC_CP16(laddr, gaddr, 112);
}

// ---------------- kernel 0: zero routing metadata ----------------
__global__ void k_init(int* counts, int* cursor, int* tok) {
    int i = blockIdx.x * blockDim.x + threadIdx.x;
    if (i < E_NUM) { counts[i] = 0; cursor[i] = 0; }
    if (i < RMAX) tok[i] = 0;
}

// ---------------- kernel 1: fp32 -> bf16 conversion of x, W1, W2 ----------------
__global__ __launch_bounds__(256) void k_convert(
        const float* __restrict__ x, const float* __restrict__ W1,
        const float* __restrict__ W2, unsigned short* __restrict__ xb,
        unsigned short* __restrict__ w1b, unsigned short* __restrict__ w2b) {
    const long long XN = (long long)N_TOK * D_DIM / 4;
    const long long WN = (long long)E_NUM * H_DIM * D_DIM / 4;
    long long i = (long long)blockIdx.x * blockDim.x + threadIdx.x;
    const float* src;
    unsigned short* dst;
    long long j;
    if (i < XN)               { src = x;  dst = xb;  j = i; }
    else if (i < XN + WN)     { src = W1; dst = w1b; j = i - XN; }
    else if (i < XN + 2 * WN) { src = W2; dst = w2b; j = i - XN - WN; }
    else return;
    float4 v = ((const float4*)src)[j];
    ushort4 o;
    o.x = f2bf(v.x); o.y = f2bf(v.y); o.z = f2bf(v.z); o.w = f2bf(v.w);
    ((ushort4*)dst)[j] = o;
}

// ---------------- kernel 2: router (one wave32 per token) ----------------
__global__ __launch_bounds__(256) void k_router(
        const float* __restrict__ x, const float* __restrict__ Wr,
        const float* __restrict__ br, int* __restrict__ topidx,
        float* __restrict__ gates, int* __restrict__ counts) {
    int lane = threadIdx.x & 31;
    int wv   = threadIdx.x >> 5;
    int n    = blockIdx.x * 8 + wv;
    float acc[E_NUM];
#pragma unroll
    for (int e = 0; e < E_NUM; ++e) acc[e] = 0.f;
    const float4* xp = (const float4*)(x + (size_t)n * D_DIM);
#pragma unroll
    for (int i = 0; i < 8; ++i) {
        int c = i * 32 + lane;
        float4 xv = xp[c];
#pragma unroll
        for (int e = 0; e < E_NUM; ++e) {
            float4 w = ((const float4*)(Wr + (size_t)e * D_DIM))[c];
            acc[e] += xv.x * w.x + xv.y * w.y + xv.z * w.z + xv.w * w.w;
        }
    }
#pragma unroll
    for (int e = 0; e < E_NUM; ++e)
#pragma unroll
        for (int off = 16; off > 0; off >>= 1)
            acc[e] += __shfl_xor(acc[e], off, 32);
    if (lane == 0) {
        float v0 = -3.4e38f, v1 = -3.4e38f;
        int i0 = 0, i1 = 0;
#pragma unroll
        for (int e = 0; e < E_NUM; ++e) {
            float l = acc[e] + br[e];
            if (l > v0)      { v1 = v0; i1 = i0; v0 = l; i0 = e; }
            else if (l > v1) { v1 = l; i1 = e; }
        }
        float e1 = __expf(v1 - v0);
        float s  = 1.f + e1;
        topidx[n * 2]     = i0;
        topidx[n * 2 + 1] = i1;
        gates[n * 2]      = 1.f / s;
        gates[n * 2 + 1]  = e1 / s;
        atomicAdd(&counts[i0], 1);
        atomicAdd(&counts[i1], 1);
    }
}

// ---------------- kernel 3: 64-aligned per-expert segment offsets ----------------
__global__ void k_offsets(const int* __restrict__ counts, int* __restrict__ offs) {
    if (threadIdx.x == 0 && blockIdx.x == 0) {
        int run = 0;
        for (int e = 0; e < E_NUM; ++e) {
            offs[e] = run;
            run += (counts[e] + 63) & ~63;
        }
        offs[E_NUM] = run;
    }
}

// ---------------- kernel 4: scatter token assignments into expert segments ----------------
__global__ __launch_bounds__(256) void k_assign(
        const int* __restrict__ topidx, int* __restrict__ cursor,
        const int* __restrict__ offs, int* __restrict__ tok, int* __restrict__ slot) {
    int n = blockIdx.x * blockDim.x + threadIdx.x;
    if (n >= N_TOK) return;
#pragma unroll
    for (int k = 0; k < TOPK; ++k) {
        int e = topidx[n * 2 + k];
        int p = atomicAdd(&cursor[e], 1);
        int s = offs[e] + p;
        tok[s] = n;
        slot[n * 2 + k] = s;
    }
}

// ---------------- kernel 5: GEMM1  h = relu(Xg @ W1[e]^T + b1[e])  (bf16 WMMA) ----
// grid: (H/256, RMAX/64), block 256 (8 waves). 64 rows x 256 cols per block.
// Double-buffered LDS staging via CDNA5 async global->LDS copies (ASYNCcnt).
__global__ __launch_bounds__(256) void k_ffn1(
        const unsigned short* __restrict__ xb, const unsigned short* __restrict__ w1b,
        const float* __restrict__ b1, const int* __restrict__ offs,
        const int* __restrict__ tok, unsigned short* __restrict__ hbuf) {
    __shared__ unsigned short sx[2][MTILE * 256];     // 2 x 32 KB chunks (64 rows x 256 K)
    const int hbase = blockIdx.x * 256;
    const int r0    = blockIdx.y * MTILE;
    int e = 0;
#pragma unroll
    for (int i = 0; i < E_NUM - 1; ++i)
        if (r0 >= offs[i + 1]) e = i + 1;

    const int lane = threadIdx.x & 31, wv = threadIdx.x >> 5;
    const int srow = threadIdx.x >> 2, ssub = threadIdx.x & 3;   // staging: 4 threads/row
    const int m    = lane & 15;
    const int half = (lane < 16) ? 0 : 8;             // A-fragment K base
    const int kb   = (lane < 16) ? 0 : 8;             // B-fragment K base
    const int trow = tok[r0 + srow];

    const unsigned long long gsrc =
        (unsigned long long)(uintptr_t)(xb + (size_t)trow * D_DIM + ssub * 64);
    const unsigned lbase0 = (unsigned)(uintptr_t)(&sx[0][srow * 256 + ssub * 64]);
    const unsigned lbase1 = (unsigned)(uintptr_t)(&sx[1][srow * 256 + ssub * 64]);

    v8f acc[8];
#pragma unroll
    for (int t = 0; t < 8; ++t)
#pragma unroll
        for (int i = 0; i < 8; ++i) acc[t][i] = 0.f;

    const unsigned short* w1e = w1b + (size_t)e * H_DIM * D_DIM;
    const int NCH = D_DIM / 256;                      // 4 chunks of K=256

    async_stage_row128(lbase0, gsrc);                 // prefetch chunk 0
    for (int ch = 0; ch < NCH; ++ch) {
        if (ch + 1 < NCH) {
            async_stage_row128((ch + 1) & 1 ? lbase1 : lbase0,
                               gsrc + (size_t)(ch + 1) * 512);
            asm volatile("s_wait_asynccnt 8" ::: "memory");   // chunk ch landed
        } else {
            asm volatile("s_wait_asynccnt 0" ::: "memory");
        }
        __syncthreads();                              // all waves' stores visible
        const unsigned short* sb = sx[ch & 1];
        for (int kk = 0; kk < 256; kk += 32) {
            AF a[4], b0, b1f;
#pragma unroll
            for (int rt = 0; rt < 4; ++rt) {
                a[rt].u4[0] = *(const uint4*)(sb + (rt * 16 + m) * 256 + kk + half);
                a[rt].u4[1] = *(const uint4*)(sb + (rt * 16 + m) * 256 + kk + 16 + half);
            }
            const int gk = ch * 256 + kk + kb;
            const unsigned short* p0 = w1e + (size_t)(hbase + (wv * 2 + 0) * 16 + m) * D_DIM + gk;
            const unsigned short* p1 = w1e + (size_t)(hbase + (wv * 2 + 1) * 16 + m) * D_DIM + gk;
            b0.u4[0]  = *(const uint4*)p0;  b0.u4[1]  = *(const uint4*)(p0 + 16);
            b1f.u4[0] = *(const uint4*)p1;  b1f.u4[1] = *(const uint4*)(p1 + 16);
#pragma unroll
            for (int rt = 0; rt < 4; ++rt) {
                acc[rt * 2 + 0] = __builtin_amdgcn_wmma_f32_16x16x32_bf16(
                    false, a[rt].v, false, b0.v,  (short)0, acc[rt * 2 + 0], false, false);
                acc[rt * 2 + 1] = __builtin_amdgcn_wmma_f32_16x16x32_bf16(
                    false, a[rt].v, false, b1f.v, (short)0, acc[rt * 2 + 1], false, false);
            }
        }
        __syncthreads();                              // done with this buffer
    }

    const int rowadd = (lane < 16) ? 0 : 8;
#pragma unroll
    for (int t = 0; t < 8; ++t) {
        const int rt  = (t >> 1) * 16, ct = t & 1;
        const int col = hbase + (wv * 2 + ct) * 16 + m;
        const float bias = b1[e * H_DIM + col];
#pragma unroll
        for (int i = 0; i < 8; ++i) {
            const int row = r0 + rt + rowadd + i;
            float v = acc[t][i] + bias;
            v = v > 0.f ? v : 0.f;                    // ReLU
            hbuf[(size_t)row * H_DIM + col] = f2bf(v);
        }
    }
}

// ---------------- kernel 6: GEMM2  y = h @ W2[e]^T + b2[e]  (bf16 WMMA) ----------
// grid: (D/256, RMAX/64), block 256.
__global__ __launch_bounds__(256) void k_ffn2(
        const unsigned short* __restrict__ hbuf, const unsigned short* __restrict__ w2b,
        const float* __restrict__ b2, const int* __restrict__ offs,
        float* __restrict__ ybuf) {
    __shared__ unsigned short sh[2][MTILE * 256];
    const int dbase = blockIdx.x * 256;
    const int r0    = blockIdx.y * MTILE;
    int e = 0;
#pragma unroll
    for (int i = 0; i < E_NUM - 1; ++i)
        if (r0 >= offs[i + 1]) e = i + 1;

    const int lane = threadIdx.x & 31, wv = threadIdx.x >> 5;
    const int srow = threadIdx.x >> 2, ssub = threadIdx.x & 3;
    const int m    = lane & 15;
    const int half = (lane < 16) ? 0 : 8;
    const int kb   = (lane < 16) ? 0 : 8;

    const unsigned long long gsrc =
        (unsigned long long)(uintptr_t)(hbuf + (size_t)(r0 + srow) * H_DIM + ssub * 64);
    const unsigned lbase0 = (unsigned)(uintptr_t)(&sh[0][srow * 256 + ssub * 64]);
    const unsigned lbase1 = (unsigned)(uintptr_t)(&sh[1][srow * 256 + ssub * 64]);

    v8f acc[8];
#pragma unroll
    for (int t = 0; t < 8; ++t)
#pragma unroll
        for (int i = 0; i < 8; ++i) acc[t][i] = 0.f;

    const unsigned short* w2e = w2b + (size_t)e * D_DIM * H_DIM;
    const int NCH = H_DIM / 256;                      // 16 chunks of K=256

    async_stage_row128(lbase0, gsrc);
    for (int ch = 0; ch < NCH; ++ch) {
        if (ch + 1 < NCH) {
            async_stage_row128((ch + 1) & 1 ? lbase1 : lbase0,
                               gsrc + (size_t)(ch + 1) * 512);
            asm volatile("s_wait_asynccnt 8" ::: "memory");
        } else {
            asm volatile("s_wait_asynccnt 0" ::: "memory");
        }
        __syncthreads();
        const unsigned short* sb = sh[ch & 1];
        for (int kk = 0; kk < 256; kk += 32) {
            AF a[4], b0, b1f;
#pragma unroll
            for (int rt = 0; rt < 4; ++rt) {
                a[rt].u4[0] = *(const uint4*)(sb + (rt * 16 + m) * 256 + kk + half);
                a[rt].u4[1] = *(const uint4*)(sb + (rt * 16 + m) * 256 + kk + 16 + half);
            }
            const int gk = ch * 256 + kk + kb;
            const unsigned short* p0 = w2e + (size_t)(dbase + (wv * 2 + 0) * 16 + m) * H_DIM + gk;
            const unsigned short* p1 = w2e + (size_t)(dbase + (wv * 2 + 1) * 16 + m) * H_DIM + gk;
            b0.u4[0]  = *(const uint4*)p0;  b0.u4[1]  = *(const uint4*)(p0 + 16);
            b1f.u4[0] = *(const uint4*)p1;  b1f.u4[1] = *(const uint4*)(p1 + 16);
#pragma unroll
            for (int rt = 0; rt < 4; ++rt) {
                acc[rt * 2 + 0] = __builtin_amdgcn_wmma_f32_16x16x32_bf16(
                    false, a[rt].v, false, b0.v,  (short)0, acc[rt * 2 + 0], false, false);
                acc[rt * 2 + 1] = __builtin_amdgcn_wmma_f32_16x16x32_bf16(
                    false, a[rt].v, false, b1f.v, (short)0, acc[rt * 2 + 1], false, false);
            }
        }
        __syncthreads();
    }

    const int rowadd = (lane < 16) ? 0 : 8;
#pragma unroll
    for (int t = 0; t < 8; ++t) {
        const int rt  = (t >> 1) * 16, ct = t & 1;
        const int col = dbase + (wv * 2 + ct) * 16 + m;
        const float bias = b2[e * D_DIM + col];
#pragma unroll
        for (int i = 0; i < 8; ++i) {
            const int row = r0 + rt + rowadd + i;
            ybuf[(size_t)row * D_DIM + col] = acc[t][i] + bias;
        }
    }
}

// ---------------- kernel 7: gated combine (deterministic, no atomics) -----------
__global__ __launch_bounds__(256) void k_combine(
        const float* __restrict__ ybuf, const float* __restrict__ gates,
        const int* __restrict__ slot, float* __restrict__ out) {
    int idx = blockIdx.x * blockDim.x + threadIdx.x;   // over N*D/4 float4s
    int n   = idx >> 8;                                // D/4 = 256 per token
    int d4  = idx & 255;
    int   s0 = slot[n * 2], s1 = slot[n * 2 + 1];
    float g0 = gates[n * 2], g1 = gates[n * 2 + 1];
    float4 y0 = ((const float4*)(ybuf + (size_t)s0 * D_DIM))[d4];
    float4 y1 = ((const float4*)(ybuf + (size_t)s1 * D_DIM))[d4];
    float4 o;
    o.x = g0 * y0.x + g1 * y1.x;
    o.y = g0 * y0.y + g1 * y1.y;
    o.z = g0 * y0.z + g1 * y1.z;
    o.w = g0 * y0.w + g1 * y1.w;
    ((float4*)out)[idx] = o;
}

// ---------------- host-side launch ----------------
extern "C" void kernel_launch(void* const* d_in, const int* in_sizes, int n_in,
                              void* d_out, int out_size, void* d_ws, size_t ws_size,
                              hipStream_t stream) {
    const float* x  = (const float*)d_in[0];
    const float* Wr = (const float*)d_in[1];
    const float* br = (const float*)d_in[2];
    const float* W1 = (const float*)d_in[3];
    const float* b1 = (const float*)d_in[4];
    const float* W2 = (const float*)d_in[5];
    const float* b2 = (const float*)d_in[6];
    float* out = (float*)d_out;
    (void)in_sizes; (void)n_in; (void)out_size; (void)ws_size;

    char* ws = (char*)d_ws;
    constexpr size_t XB_OFF  = 0;
    constexpr size_t W1B_OFF = XB_OFF  + (size_t)N_TOK * D_DIM * 2;          //  +8 MiB
    constexpr size_t W2B_OFF = W1B_OFF + (size_t)E_NUM * H_DIM * D_DIM * 2;  // +64 MiB
    constexpr size_t HB_OFF  = W2B_OFF + (size_t)E_NUM * D_DIM * H_DIM * 2;  // +64 MiB
    constexpr size_t YB_OFF  = HB_OFF  + (size_t)RMAX * H_DIM * 2;           // +68 MiB
    constexpr size_t MT_OFF  = YB_OFF  + (size_t)RMAX * D_DIM * 4;           // +34 MiB

    unsigned short* xb   = (unsigned short*)(ws + XB_OFF);
    unsigned short* w1b  = (unsigned short*)(ws + W1B_OFF);
    unsigned short* w2b  = (unsigned short*)(ws + W2B_OFF);
    unsigned short* hbuf = (unsigned short*)(ws + HB_OFF);
    float*          ybuf = (float*)(ws + YB_OFF);
    int*   counts = (int*)(ws + MT_OFF);
    int*   cursor = counts + 8;
    int*   offs   = cursor + 8;
    int*   topidx = offs + 16;
    float* gates  = (float*)(topidx + N_TOK * 2);
    int*   slot   = (int*)(gates + N_TOK * 2);
    int*   tok    = slot + N_TOK * 2;

    k_init<<<(RMAX + 255) / 256, 256, 0, stream>>>(counts, cursor, tok);

    const long long cvt4 = (long long)N_TOK * D_DIM / 4 + 2LL * E_NUM * H_DIM * D_DIM / 4;
    k_convert<<<(unsigned)((cvt4 + 255) / 256), 256, 0, stream>>>(x, W1, W2, xb, w1b, w2b);

    k_router<<<N_TOK / 8, 256, 0, stream>>>(x, Wr, br, topidx, gates, counts);
    k_offsets<<<1, 32, 0, stream>>>(counts, offs);
    k_assign<<<N_TOK / 256, 256, 0, stream>>>(topidx, cursor, offs, tok, slot);

    k_ffn1<<<dim3(H_DIM / 256, RMAX / MTILE), 256, 0, stream>>>(xb, w1b, b1, offs, tok, hbuf);
    k_ffn2<<<dim3(D_DIM / 256, RMAX / MTILE), 256, 0, stream>>>(hbuf, w2b, b2, offs, ybuf);

    k_combine<<<(N_TOK * D_DIM / 4) / 256, 256, 0, stream>>>(ybuf, gates, slot, out);
}
